// KernelLayer_41875931136544
// MI455X (gfx1250) — matrix operand order repaired and checked by reading, not compiled, not run on hardware
//
#include <hip/hip_runtime.h>
#include <math.h>

#define BS 8192
#define DD 128

typedef __attribute__((ext_vector_type(16))) __bf16 v16bf;
typedef __attribute__((ext_vector_type(8)))  float  v8f;
typedef __attribute__((ext_vector_type(8)))  unsigned short us8;

union V16 { v16bf bf; us8 h[2]; };

__device__ __forceinline__ unsigned short bf16_rne(float f) {
    unsigned int u = __float_as_uint(f);
    unsigned int r = u + 0x7fffu + ((u >> 16) & 1u);
    return (unsigned short)(r >> 16);
}

// --- Pass 0: exact fp32 row norms + bf16 hi/lo split of x -------------------
__global__ void __launch_bounds__(256) prep_kernel(
    const float* __restrict__ x, float* __restrict__ sq,
    unsigned short* __restrict__ xhi, unsigned short* __restrict__ xlo)
{
    int wave = threadIdx.x >> 5, lane = threadIdx.x & 31;
    int row  = blockIdx.x * 8 + wave;
    float4 v = ((const float4*)(x + (size_t)row * DD))[lane];
    float fs[4] = {v.x, v.y, v.z, v.w};
    unsigned short h[4], l[4];
    float s = 0.0f;
#pragma unroll
    for (int i = 0; i < 4; ++i) {
        unsigned int u = __float_as_uint(fs[i]);
        h[i] = (unsigned short)(u >> 16);                 // truncated hi
        float hf = __uint_as_float(u & 0xffff0000u);
        l[i] = bf16_rne(fs[i] - hf);                      // RNE lo
        s += fs[i] * fs[i];
    }
    ushort4 hv; hv.x=h[0]; hv.y=h[1]; hv.z=h[2]; hv.w=h[3];
    ushort4 lv; lv.x=l[0]; lv.y=l[1]; lv.z=l[2]; lv.w=l[3];
    unsigned o = (unsigned)row * DD + lane * 4;
    *(ushort4*)(xhi + o) = hv;
    *(ushort4*)(xlo + o) = lv;
#pragma unroll
    for (int off = 16; off >= 1; off >>= 1) s += __shfl_xor(s, off, 32);
    if (lane == 0) sq[row] = s;
}

// --- Pass 1: symmetric tiled WMMA GEMM + exp + partial row sums -------------
__global__ void __launch_bounds__(256) rbf_gemm(
    const unsigned short* __restrict__ xhi,
    const unsigned short* __restrict__ xlo,
    const float* __restrict__ sq,
    float* __restrict__ Kout,
    float* __restrict__ partial)  // [64][8192]
{
    __shared__ float ldsT[8][16 * 17];   // per-wave transpose staging
    __shared__ float colpart[8][128];    // per-wave column partial sums

    // triangular block decode: t -> (bi >= bj)
    int t = blockIdx.x;
    int bi = (int)((sqrtf(8.0f * (float)t + 1.0f) - 1.0f) * 0.5f);
    while ((bi + 1) * (bi + 2) / 2 <= t) ++bi;
    while (bi * (bi + 1) / 2 > t) --bi;
    int bj = t - bi * (bi + 1) / 2;
    int bm = bi * 128, bn = bj * 128;
    bool offdiag = (bi != bj);

    int w = threadIdx.x >> 5, lane = threadIdx.x & 31;
    int l15 = lane & 15, lh = lane >> 4;

    // 32-bit element offsets against uniform SGPR bases (saddr+voffset form).
    // A operand: row = bm + w*16 + l15; chunks at K = k0 + lh*8 and +16.
    unsigned aoff = (unsigned)(bm + w * 16 + l15) * DD + (unsigned)(lh * 8);
    // B operand: row = bn + wn*16 + l15; chunks at K = k0 + lh*8 and +16.
    unsigned boff[8];
#pragma unroll
    for (int wn = 0; wn < 8; ++wn)
        boff[wn] = (unsigned)(bn + wn * 16 + l15) * DD + (unsigned)(lh * 8);

    v8f zero = {0.f,0.f,0.f,0.f,0.f,0.f,0.f,0.f};
    v8f acc[8];
#pragma unroll
    for (int wn = 0; wn < 8; ++wn) acc[wn] = zero;

#pragma unroll
    for (int k0 = 0; k0 < DD; k0 += 32) {
        V16 ahi, alo;
        ahi.h[0] = *(const us8*)(xhi + aoff + k0);
        ahi.h[1] = *(const us8*)(xhi + aoff + k0 + 16);
        alo.h[0] = *(const us8*)(xlo + aoff + k0);
        alo.h[1] = *(const us8*)(xlo + aoff + k0 + 16);
#pragma unroll
        for (int wn = 0; wn < 8; ++wn) {
            V16 bhi, blo;
            bhi.h[0] = *(const us8*)(xhi + boff[wn] + k0);
            bhi.h[1] = *(const us8*)(xhi + boff[wn] + k0 + 16);
            blo.h[0] = *(const us8*)(xlo + boff[wn] + k0);
            blo.h[1] = *(const us8*)(xlo + boff[wn] + k0 + 16);
            v8f c = acc[wn];
            // fp32-accurate product: hi*hi + hi*lo + lo*hi
            c = __builtin_amdgcn_wmma_f32_16x16x32_bf16(false, alo.bf, false, bhi.bf, (short)0, c, false, false);
            c = __builtin_amdgcn_wmma_f32_16x16x32_bf16(false, ahi.bf, false, blo.bf, (short)0, c, false, false);
            c = __builtin_amdgcn_wmma_f32_16x16x32_bf16(false, ahi.bf, false, bhi.bf, (short)0, c, false, false);
            acc[wn] = c;
        }
    }

    // ---- epilogue: exp, stores (direct + mirrored), partial sums ----
    const float inv_neg = -1.0f / 512.0f;     // -1/(2*sigma^2), sigma=16
    int gr0 = bm + w * 16 + lh * 8;           // first row this lane holds
    float sqr[8];
#pragma unroll
    for (int r = 0; r < 8; ++r) sqr[r] = sq[gr0 + r];
    float rs[8];
#pragma unroll
    for (int r = 0; r < 8; ++r) rs[r] = 0.0f;

#pragma unroll
    for (int wn = 0; wn < 8; ++wn) {
        int gc = bn + wn * 16 + l15;
        float sqc = sq[gc];
        float cs = 0.0f;
        float kv[8];
#pragma unroll
        for (int r = 0; r < 8; ++r) {
            float d2 = sqr[r] + sqc - 2.0f * acc[wn][r];
            d2 = fmaxf(d2, 0.0f);
            float k = __expf(d2 * inv_neg);
            kv[r] = k; rs[r] += k; cs += k;
            Kout[(unsigned)(gr0 + r) * BS + (unsigned)gc] = k;
        }
        if (offdiag) {
            // column partials (become row sums of the mirrored block rows)
            cs += __shfl_xor(cs, 16, 32);
            if (lane < 16) colpart[w][wn * 16 + l15] = cs;
            // mirrored tile via per-wave LDS transpose (LDS in-order per wave)
            float* lt = ldsT[w];
#pragma unroll
            for (int r = 0; r < 8; ++r)
                lt[(lh * 8 + r) * 17 + l15] = kv[r];
            float tv[8];
#pragma unroll
            for (int r = 0; r < 8; ++r)
                tv[r] = lt[l15 * 17 + lh * 8 + r];
            unsigned tbase = (unsigned)(bn + wn * 16 + lh * 8) * BS
                           + (unsigned)(bm + w * 16 + l15);
#pragma unroll
            for (int r = 0; r < 8; ++r)
                Kout[tbase + (unsigned)r * BS] = tv[r];
        }
    }

    // deterministic per-(rowblock, colblock) row partials
#pragma unroll
    for (int r = 0; r < 8; ++r) {
        float v = rs[r];
        v += __shfl_xor(v, 1, 32); v += __shfl_xor(v, 2, 32);
        v += __shfl_xor(v, 4, 32); v += __shfl_xor(v, 8, 32);
        if (l15 == 0) partial[(unsigned)bj * BS + (unsigned)(gr0 + r)] = v;
    }
    if (offdiag) {
        __syncthreads();
        int tt = threadIdx.x;
        if (tt < 128) {
            float s = 0.0f;
#pragma unroll
            for (int ww = 0; ww < 8; ++ww) s += colpart[ww][tt];
            partial[(unsigned)bi * BS + (unsigned)(bn + tt)] = s;
        }
    }
}

// --- Pass 2: ds = 1/sqrt(rowsum) --------------------------------------------
__global__ void __launch_bounds__(256) ds_kernel(
    const float* __restrict__ partial, float* __restrict__ ds)
{
    int i = blockIdx.x * 256 + threadIdx.x;
    float s = 0.0f;
#pragma unroll 8
    for (int cb = 0; cb < 64; ++cb) s += partial[(unsigned)cb * BS + (unsigned)i];
    ds[i] = 1.0f / sqrtf(s);
}

// --- Pass 3: out *= ds_i * ds_j ---------------------------------------------
__global__ void __launch_bounds__(256) scale_kernel(
    float* __restrict__ out, const float* __restrict__ ds)
{
    unsigned t = blockIdx.x * 256 + threadIdx.x;   // float4 index, < 2^24
    int i  = (int)(t >> 11);          // t / (8192/4)
    int j4 = (int)(t & 2047);
    float  di = ds[i];
    float4 dj = ((const float4*)ds)[j4];
    float4* p = (float4*)out + t;
    float4 v = *p;
    v.x *= di * dj.x; v.y *= di * dj.y; v.z *= di * dj.z; v.w *= di * dj.w;
    *p = v;
}

extern "C" void kernel_launch(void* const* d_in, const int* in_sizes, int n_in,
                              void* d_out, int out_size, void* d_ws, size_t ws_size,
                              hipStream_t stream) {
    const float* x = (const float*)d_in[0];
    float* out = (float*)d_out;

    // workspace layout (≈6.1 MB)
    float* sq      = (float*)d_ws;             // 8192 f32
    float* ds      = sq + BS;                  // 8192 f32
    float* partial = ds + BS;                  // 64*8192 f32
    unsigned short* xhi = (unsigned short*)(partial + 64 * (size_t)BS); // 1M bf16
    unsigned short* xlo = xhi + (size_t)BS * DD;                        // 1M bf16

    prep_kernel <<<BS / 8, 256, 0, stream>>>(x, sq, xhi, xlo);
    rbf_gemm    <<<64 * 65 / 2, 256, 0, stream>>>(xhi, xlo, sq, out, partial);
    ds_kernel   <<<BS / 256, 256, 0, stream>>>(partial, ds);
    scale_kernel<<<(BS * (BS / 4)) / 256, 256, 0, stream>>>(out, ds);
}
